// _GraphConvPyTorchModel_90864328114348
// MI455X (gfx1250) — matrix operand order repaired and checked by reading, not compiled, not run on hardware
//
#include <hip/hip_runtime.h>
#include <math.h>

#define NATOMS 200000

typedef __attribute__((ext_vector_type(16))) __bf16        v16bf;
typedef __attribute__((ext_vector_type(8)))  float         v8f;
typedef __attribute__((ext_vector_type(8)))  unsigned int  v8u;

__device__ __forceinline__ unsigned short f2bf(float f) {
  unsigned int u = __builtin_bit_cast(unsigned int, f);
  u = (u + 0x7fffu + ((u >> 16) & 1u)) >> 16;   // round-to-nearest-even
  return (unsigned short)u;
}

// Build a 16-element bf16 fragment (8 VGPRs) from two 16-byte LDS chunks.
__device__ __forceinline__ v16bf ld_frag(const unsigned short* p0, const unsigned short* p1) {
  uint4 a = *(const uint4*)p0;
  uint4 b = *(const uint4*)p1;
  v8u r = { a.x, a.y, a.z, a.w, b.x, b.y, b.z, b.w };
  return __builtin_bit_cast(v16bf, r);
}

#define BNS 0.99999500003749968f  // 1/sqrt(1+1e-5), eval-mode BN scale

// ---------------------------------------------------------------------------
// Graph conv + ReLU + BN. One wave computes a 16-atom x 128-out tile.
// Weights for the self pass and the neighbor-sum pass are staged (transposed,
// bf16) into shared LDS one at a time; accumulators persist in VGPRs.
// ---------------------------------------------------------------------------
template <int KIN, int KP, int NW>
__global__ __launch_bounds__(32 * NW)
void conv_bn_kernel(const float* __restrict__ x, float* __restrict__ y,
                    const float* __restrict__ Wall, const float* __restrict__ ball,
                    const float* __restrict__ gamma, const float* __restrict__ beta,
                    const int* __restrict__ adj, int d, int off, int cnt) {
  __shared__ __attribute__((aligned(16))) unsigned short WT[128 * KP];        // [n][k] bf16
  __shared__ __attribute__((aligned(16))) unsigned short AS[NW][2][16 * KP];  // per-wave A tiles

  const int tid  = threadIdx.x;
  const int wave = tid >> 5;
  const int lane = tid & 31;
  const int rbase = off + (blockIdx.x * NW + wave) * 16;
  const int lim   = off + cnt;
  const int ncol  = lane & 15;

  // ---- stage A tiles: [1]=self features, [0]=neighbor sum ----
  unsigned short* Asum = &AS[wave][0][0];
  unsigned short* Aslf = &AS[wave][1][0];
  for (int idx = lane; idx < 16 * KP; idx += 32) {
    const int r = idx / KP;
    const int k = idx - r * KP;
    const int atom = rbase + r;
    float vself = 0.f, vsum = 0.f;
    if (atom < lim && k < KIN) {
      vself = x[atom * KIN + k];
      if (d >= 1) {
        const int local = atom - off;
        for (int j = 0; j < d; ++j)
          vsum += x[adj[local * d + j] * KIN + k];
      }
    }
    Aslf[idx] = f2bf(vself);
    Asum[idx] = f2bf(vsum);
  }

  const int wn = (d >= 1) ? 2 * (d - 1) : -1;  // neighbor-sum weights
  const int ws = (d >= 1) ? 2 * d - 1 : 12;    // self weights

  v8f acc[8];
#pragma unroll
  for (int nt = 0; nt < 8; ++nt) {
    const int n = nt * 16 + ncol;
    float bv = ball[ws * 128 + n];
    if (wn >= 0) bv += ball[wn * 128 + n];
    acc[nt] = (v8f){bv, bv, bv, bv, bv, bv, bv, bv};
  }

  for (int pass = 0; pass < 2; ++pass) {
    const int widx = (pass == 0) ? ws : wn;
    if (widx < 0) break;  // uniform across the block
    __syncthreads();
    const float* W = Wall + widx * KIN * 128;
    for (int idx = tid; idx < 128 * KP; idx += 32 * NW) {
      const int n = idx / KP;
      const int k = idx - n * KP;
      WT[idx] = f2bf((k < KIN) ? W[k * 128 + n] : 0.f);
    }
    __syncthreads();
    const unsigned short* A = (pass == 0) ? Aslf : Asum;
#pragma unroll
    for (int ks = 0; ks < KP / 32; ++ks) {
      const int kb  = ks * 32;
      const int ao  = kb + ((lane < 16) ? 0 : 8);   // A: k={kb..+7,kb+16..+23} | {kb+8..,kb+24..}
      const int kbb = kb + ((lane < 16) ? 0 : 8);   // B: k={kb..+7,kb+16..+23} | {kb+8..,kb+24..}
      v16bf af = ld_frag(A + ncol * KP + ao, A + ncol * KP + ao + 16);
#pragma unroll
      for (int nt = 0; nt < 8; ++nt) {
        const int n = nt * 16 + ncol;
        v16bf bfr = ld_frag(WT + n * KP + kbb, WT + n * KP + kbb + 16);
        acc[nt] = __builtin_amdgcn_wmma_f32_16x16x32_bf16(
            false, af, false, bfr, (short)0, acc[nt], false, false);
      }
    }
  }

  // ---- ReLU + BN + store ----
#pragma unroll
  for (int nt = 0; nt < 8; ++nt) {
    const int n = nt * 16 + ncol;
    const float gm = gamma[n] * BNS;
    const float bt = beta[n];
#pragma unroll
    for (int gi = 0; gi < 8; ++gi) {
      const int atom = rbase + gi + ((lane < 16) ? 0 : 8);
      if (atom < lim) {
        float v = acc[nt][gi];
        v = (v > 0.f) ? v : 0.f;
        y[atom * 128 + n] = v * gm + bt;
      }
    }
  }
}

// ---------------------------------------------------------------------------
// graph_pool: out[i] = max(self, neighbors) per feature. float4 per thread.
// ---------------------------------------------------------------------------
__global__ __launch_bounds__(256)
void pool_kernel(const float* __restrict__ x, float* __restrict__ y,
                 const int* __restrict__ a1, const int* __restrict__ a2,
                 const int* __restrict__ a3, const int* __restrict__ a4,
                 const int* __restrict__ a5, const int* __restrict__ a6) {
  const int gid  = blockIdx.x * 256 + threadIdx.x;
  const int atom = gid >> 5;
  const int f    = (gid & 31) * 4;
  if (atom >= NATOMS) return;
  float4 m = *(const float4*)(x + atom * 128 + f);
  int d = 0, local = 0;
  const int* adj = a1;
  if (atom < 2000)        { d = 0; }
  else if (atom < 42000)  { d = 1; local = atom - 2000;   adj = a1; }
  else if (atom < 102000) { d = 2; local = atom - 42000;  adj = a2; }
  else if (atom < 160000) { d = 3; local = atom - 102000; adj = a3; }
  else if (atom < 198000) { d = 4; local = atom - 160000; adj = a4; }
  else if (atom < 199500) { d = 5; local = atom - 198000; adj = a5; }
  else                    { d = 6; local = atom - 199500; adj = a6; }
  for (int j = 0; j < d; ++j) {
    const int nb = adj[local * d + j];
    float4 v = *(const float4*)(x + nb * 128 + f);
    m.x = fmaxf(m.x, v.x); m.y = fmaxf(m.y, v.y);
    m.z = fmaxf(m.z, v.z); m.w = fmaxf(m.w, v.w);
  }
  *(float4*)(y + atom * 128 + f) = m;
}

// ---------------------------------------------------------------------------
// Fused dense0+BN2 -> dense1+BN3 -> segment-sum (atomic) per 16-row tile.
// ---------------------------------------------------------------------------
template <int NW>
__global__ __launch_bounds__(32 * NW)
void dense_gather_kernel(const float* __restrict__ x,
                         const float* __restrict__ W0, const float* __restrict__ b0,
                         const float* __restrict__ W1, const float* __restrict__ b1,
                         const float* __restrict__ g2, const float* __restrict__ be2,
                         const float* __restrict__ g3, const float* __restrict__ be3,
                         const int* __restrict__ mem, float* __restrict__ fp_pre) {
  __shared__ __attribute__((aligned(16))) unsigned short WT0[64 * 128];   // [n][k]
  __shared__ __attribute__((aligned(16))) unsigned short WT1[64 * 64];    // [n][k]
  __shared__ __attribute__((aligned(16))) unsigned short AS[NW][16 * 128];
  __shared__ __attribute__((aligned(16))) unsigned short HS[NW][16 * 64];

  const int tid = threadIdx.x, wave = tid >> 5, lane = tid & 31;
  const int rbase = (blockIdx.x * NW + wave) * 16;  // NATOMS % (16*NW) == 0
  const int ncol  = lane & 15;

  for (int idx = tid; idx < 64 * 128; idx += 32 * NW) {
    const int n = idx >> 7, k = idx & 127;
    WT0[idx] = f2bf(W0[k * 64 + n]);
  }
  for (int idx = tid; idx < 64 * 64; idx += 32 * NW) {
    const int n = idx >> 6, k = idx & 63;
    WT1[idx] = f2bf(W1[k * 64 + n]);
  }
  unsigned short* A = &AS[wave][0];
  for (int idx = lane; idx < 16 * 128; idx += 32) {
    const int r = idx >> 7, k = idx & 127;
    A[idx] = f2bf(x[(rbase + r) * 128 + k]);
  }
  __syncthreads();

  // ---- GEMM 1: [16,128] x [128,64] ----
  v8f acc[4];
#pragma unroll
  for (int nt = 0; nt < 4; ++nt) {
    const float bv = b0[nt * 16 + ncol];
    acc[nt] = (v8f){bv, bv, bv, bv, bv, bv, bv, bv};
  }
#pragma unroll
  for (int ks = 0; ks < 4; ++ks) {
    const int kb  = ks * 32;
    const int ao  = kb + ((lane < 16) ? 0 : 8);
    const int kbb = kb + ((lane < 16) ? 0 : 8);
    v16bf af = ld_frag(A + ncol * 128 + ao, A + ncol * 128 + ao + 16);
#pragma unroll
    for (int nt = 0; nt < 4; ++nt) {
      const int n = nt * 16 + ncol;
      v16bf bfr = ld_frag(WT0 + n * 128 + kbb, WT0 + n * 128 + kbb + 16);
      acc[nt] = __builtin_amdgcn_wmma_f32_16x16x32_bf16(
          false, af, false, bfr, (short)0, acc[nt], false, false);
    }
  }
  // epilogue 0: ReLU+BN2 -> bf16 to LDS
  unsigned short* H = &HS[wave][0];
#pragma unroll
  for (int nt = 0; nt < 4; ++nt) {
    const int n = nt * 16 + ncol;
    const float gm = g2[n] * BNS, bt = be2[n];
#pragma unroll
    for (int gi = 0; gi < 8; ++gi) {
      const int row = gi + ((lane < 16) ? 0 : 8);
      float v = acc[nt][gi];
      v = (v > 0.f) ? v : 0.f;
      H[row * 64 + n] = f2bf(v * gm + bt);
    }
  }
  __syncthreads();

  // ---- GEMM 2: [16,64] x [64,64] ----
  v8f acc1[4];
#pragma unroll
  for (int nt = 0; nt < 4; ++nt) {
    const float bv = b1[nt * 16 + ncol];
    acc1[nt] = (v8f){bv, bv, bv, bv, bv, bv, bv, bv};
  }
#pragma unroll
  for (int ks = 0; ks < 2; ++ks) {
    const int kb  = ks * 32;
    const int ao  = kb + ((lane < 16) ? 0 : 8);
    const int kbb = kb + ((lane < 16) ? 0 : 8);
    v16bf af = ld_frag(H + ncol * 64 + ao, H + ncol * 64 + ao + 16);
#pragma unroll
    for (int nt = 0; nt < 4; ++nt) {
      const int n = nt * 16 + ncol;
      v16bf bfr = ld_frag(WT1 + n * 64 + kbb, WT1 + n * 64 + kbb + 16);
      acc1[nt] = __builtin_amdgcn_wmma_f32_16x16x32_bf16(
          false, af, false, bfr, (short)0, acc1[nt], false, false);
    }
  }
  // epilogue 1: ReLU+BN3 -> atomic segment sum
#pragma unroll
  for (int nt = 0; nt < 4; ++nt) {
    const int n = nt * 16 + ncol;
    const float gm = g3[n] * BNS, bt = be3[n];
#pragma unroll
    for (int gi = 0; gi < 8; ++gi) {
      const int atom = rbase + gi + ((lane < 16) ? 0 : 8);
      float v = acc1[nt][gi];
      v = (v > 0.f) ? v : 0.f;
      atomicAdd(&fp_pre[mem[atom] * 64 + n], v * gm + bt);
    }
  }
}

__global__ __launch_bounds__(256)
void zero_kernel(float* __restrict__ p, int n) {
  const int i = blockIdx.x * 256 + threadIdx.x;
  if (i < n) p[i] = 0.f;
}

// tanh + fp store + regression / uncertainty heads
__global__ __launch_bounds__(64)
void finalize_kernel(const float* __restrict__ fp_pre,
                     const float* __restrict__ rW, const float* __restrict__ rb,
                     const float* __restrict__ uW, const float* __restrict__ ub,
                     float* __restrict__ out) {
  __shared__ float f[64];
  const int g = blockIdx.x, t = threadIdx.x;
  const float v = tanhf(fp_pre[g * 64 + t]);
  f[t] = v;
  out[49152 + g * 64 + t] = v;  // fp
  __syncthreads();
  if (t < 12) {
    float s = rb[t];
    for (int k = 0; k < 64; ++k) s += f[k] * rW[k * 12 + t];
    out[g * 12 + t]         = s;  // out
    out[24576 + g * 12 + t] = s;  // out (again)
  } else if (t < 24) {
    const int tt = t - 12;
    float s = ub[tt];
    for (int k = 0; k < 64; ++k) s += f[k] * uW[k * 12 + tt];
    out[36864 + g * 12 + tt] = s;        // log_var
    out[12288 + g * 12 + tt] = expf(s);  // var
  }
}

extern "C" void kernel_launch(void* const* d_in, const int* in_sizes, int n_in,
                              void* d_out, int out_size, void* d_ws, size_t ws_size,
                              hipStream_t stream) {
  (void)in_sizes; (void)n_in; (void)out_size; (void)ws_size;

  const float* atomf      = (const float*)d_in[0];
  const int*   membership = (const int*)d_in[2];
  const int*   adjs[7]    = {nullptr, (const int*)d_in[4], (const int*)d_in[5],
                             (const int*)d_in[6], (const int*)d_in[7],
                             (const int*)d_in[8], (const int*)d_in[9]};
  const float* c0W = (const float*)d_in[10];
  const float* c0b = (const float*)d_in[11];
  const float* c1W = (const float*)d_in[12];
  const float* c1b = (const float*)d_in[13];
  const float* bn0g = (const float*)d_in[14], *bn0b = (const float*)d_in[15];
  const float* bn1g = (const float*)d_in[16], *bn1b = (const float*)d_in[17];
  const float* bn2g = (const float*)d_in[18], *bn2b = (const float*)d_in[19];
  const float* bn3g = (const float*)d_in[20], *bn3b = (const float*)d_in[21];
  const float* d0W = (const float*)d_in[22], *d0b = (const float*)d_in[23];
  const float* d1W = (const float*)d_in[24], *d1b = (const float*)d_in[25];
  const float* rW  = (const float*)d_in[26], *rb  = (const float*)d_in[27];
  const float* uW  = (const float*)d_in[28], *ub  = (const float*)d_in[29];

  float* bufA   = (float*)d_ws;
  float* bufB   = bufA + (size_t)NATOMS * 128;
  float* fp_pre = bufB + (size_t)NATOMS * 128;
  float* out    = (float*)d_out;

  static const int OFF[7] = {0, 2000, 42000, 102000, 160000, 198000, 199500};
  static const int CNT[7] = {2000, 40000, 60000, 58000, 38000, 1500, 500};

  // conv0 (K=75 -> pad 96), 4 waves/block = 64 atoms/block
  for (int d = 0; d <= 6; ++d) {
    const int blocks = (CNT[d] + 63) / 64;
    conv_bn_kernel<75, 96, 4><<<blocks, 128, 0, stream>>>(
        atomf, bufA, c0W, c0b, bn0g, bn0b, adjs[d], d, OFF[d], CNT[d]);
  }
  pool_kernel<<<(NATOMS * 32 + 255) / 256, 256, 0, stream>>>(
      bufA, bufB, adjs[1], adjs[2], adjs[3], adjs[4], adjs[5], adjs[6]);

  // conv1 (K=128), 3 waves/block = 48 atoms/block (LDS fit)
  for (int d = 0; d <= 6; ++d) {
    const int blocks = (CNT[d] + 47) / 48;
    conv_bn_kernel<128, 128, 3><<<blocks, 96, 0, stream>>>(
        bufB, bufA, c1W, c1b, bn1g, bn1b, adjs[d], d, OFF[d], CNT[d]);
  }
  pool_kernel<<<(NATOMS * 32 + 255) / 256, 256, 0, stream>>>(
      bufA, bufB, adjs[1], adjs[2], adjs[3], adjs[4], adjs[5], adjs[6]);

  zero_kernel<<<(1024 * 64 + 255) / 256, 256, 0, stream>>>(fp_pre, 1024 * 64);
  dense_gather_kernel<4><<<NATOMS / 64, 128, 0, stream>>>(
      bufB, d0W, d0b, d1W, d1b, bn2g, bn2b, bn3g, bn3b, membership, fp_pre);
  finalize_kernel<<<1024, 64, 0, stream>>>(fp_pre, rW, rb, uW, ub, out);
}
